// FlashSparseAttention_6897717477932
// MI455X (gfx1250) — matrix operand order repaired and checked by reading, not compiled, not run on hardware
//
#include <hip/hip_runtime.h>

typedef __attribute__((ext_vector_type(16))) _Float16     v16h;
typedef __attribute__((ext_vector_type(8)))  float        v8f;
typedef __attribute__((ext_vector_type(4)))  unsigned int u32x4;
typedef __attribute__((ext_vector_type(8)))  int          i32x8;
typedef __attribute__((ext_vector_type(4)))  int          i32x4;

#define S_LEN 2048
#define HID   2048
#define NH    16
#define NKV   4
#define HD    128
#define KVDIM 512
#define GRP   (NH / NKV)

static __device__ __forceinline__ v8f wmma_f16(v16h a, v16h b, v8f c) {
  return __builtin_amdgcn_wmma_f32_16x16x32_f16(false, a, false, b, (short)0, c,
                                                false, false);
}

// ---------------------------------------------------------------------------
// TDM: DMA a 2-D f16 tile (rows x row_elems) from global into LDS.
// pad_code: pad_interval code (3 -> every 64B, 5 -> every 256B); pad = 8B.
// ---------------------------------------------------------------------------
static __device__ __forceinline__ void tdm_load_2d_f16(
    const _Float16* gptr, unsigned lds_off, int rows, int row_elems,
    long row_stride_elems, int pad_code)
{
  const unsigned long long ga = (unsigned long long)(uintptr_t)gptr;
  u32x4 g0;
  g0[0] = 1u;                                            // count=1, user D#
  g0[1] = lds_off;                                       // LDS byte address
  g0[2] = (unsigned)(ga & 0xFFFFFFFFu);                  // global_addr[31:0]
  g0[3] = (unsigned)((ga >> 32) & 0x1FFFFFFu) | (2u << 30); // [56:32] + type=2

  const unsigned td0 = (unsigned)row_elems;              // tensor_dim0
  const unsigned td1 = 0x40000000u;                      // tensor_dim1 (large)
  const unsigned long long st = (unsigned long long)row_stride_elems;
  i32x8 g1;
  g1[0] = (int)((1u << 16) |                             // data_size = 2 bytes
                (1u << 20) |                             // pad_enable
                ((unsigned)pad_code << 22) |             // pad_interval
                (1u << 25));                             // pad_amount = 2 DW (8B)
  g1[1] = (int)((td0 & 0xFFFFu) << 16);
  g1[2] = (int)((td0 >> 16) | ((td1 & 0xFFFFu) << 16));
  g1[3] = (int)((td1 >> 16) | ((unsigned)row_elems << 16)); // tile_dim0
  g1[4] = rows;                                          // tile_dim1, tile_dim2=0
  g1[5] = (int)(st & 0xFFFFFFFFu);                       // tensor_dim0_stride lo
  g1[6] = (int)((st >> 32) & 0xFFFFu);                   // stride hi, dim1_stride=0
  g1[7] = 0;
  i32x4 z4 = {0, 0, 0, 0};                               // 2-D: groups 2/3 unused
  i32x8 z8 = {0, 0, 0, 0, 0, 0, 0, 0};
  __builtin_amdgcn_tensor_load_to_lds(g0, g1, z4, z4, z8, 0);
}

static __device__ __forceinline__ unsigned lds_off_of(const void* p) {
  return (unsigned)(uintptr_t)p;   // LDS aperture: low 32 bits = LDS offset
}

// ---------------------------------------------------------------------------
// f32 -> f16 conversion (single pass over weights/activations)
// ---------------------------------------------------------------------------
__global__ __launch_bounds__(256)
void mi455_cvt_f16(const float* __restrict__ src, _Float16* __restrict__ dst, int n)
{
  const int i = (blockIdx.x * 256 + threadIdx.x) * 4;
  if (i < n) {
    const float4 v = *(const float4*)(src + i);
    dst[i]     = (_Float16)v.x;
    dst[i + 1] = (_Float16)v.y;
    dst[i + 2] = (_Float16)v.z;
    dst[i + 3] = (_Float16)v.w;
  }
}

// ---------------------------------------------------------------------------
// GEMM: C[M,N] = A[M,K] * W[N,K]^T  (f16 in via TDM double-buffer, fp32 out)
// ---------------------------------------------------------------------------
#define BM 128
#define BN 64
#define BK 32
#define LDT 36   // halves per LDS row: 64B data + 8B TDM pad

__global__ __launch_bounds__(256)
void mi455_gemm_f16(const _Float16* __restrict__ A, const _Float16* __restrict__ W,
                    float* __restrict__ C, int M, int N, int K)
{
  __shared__ _Float16 lsA[2][BM * LDT];
  __shared__ _Float16 lsB[2][BN * LDT];

  const int tid  = threadIdx.x;
  const int lane = tid & 31;
  const int wv   = tid >> 5;
  const int wm   = (wv & 3) * 32;
  const int wn   = (wv >> 2) * 32;
  const int g    = lane >> 4;
  const int l16  = lane & 15;

  const int mblk = blockIdx.y * BM;
  const int nblk = blockIdx.x * BN;
  (void)M;

  v8f acc00 = {}; v8f acc01 = {}; v8f acc10 = {}; v8f acc11 = {};

  const int nk = K / BK;
  if (wv == 0) {  // prologue DMA: tile 0
    tdm_load_2d_f16(A + (size_t)mblk * K, lds_off_of(&lsA[0][0]), BM, BK, K, 3);
    tdm_load_2d_f16(W + (size_t)nblk * K, lds_off_of(&lsB[0][0]), BN, BK, K, 3);
  }

  for (int kt = 0; kt < nk; ++kt) {
    if (wv == 0) __builtin_amdgcn_s_wait_tensorcnt(0);
    __syncthreads();                       // tile kt resident in buf[kt&1]
    if (kt + 1 < nk && wv == 0) {          // overlap DMA of tile kt+1
      const int k1 = (kt + 1) * BK;
      tdm_load_2d_f16(A + (size_t)mblk * K + k1, lds_off_of(&lsA[(kt + 1) & 1][0]),
                      BM, BK, K, 3);
      tdm_load_2d_f16(W + (size_t)nblk * K + k1, lds_off_of(&lsB[(kt + 1) & 1][0]),
                      BN, BK, K, 3);
    }
    const _Float16* bufA = &lsA[kt & 1][0];
    const _Float16* bufB = &lsB[kt & 1][0];

    v16h a0, a1, b0, b1;
    {
      const _Float16* pa0 = bufA + (wm + l16) * LDT;
      const _Float16* pa1 = bufA + (wm + 16 + l16) * LDT;
      #pragma unroll
      for (int j = 0; j < 8; ++j) {
        a0[j]     = pa0[g * 16 + j];
        a0[j + 8] = pa0[g * 16 + 8 + j];
        a1[j]     = pa1[g * 16 + j];
        a1[j + 8] = pa1[g * 16 + 8 + j];
      }
      const _Float16* pb0 = bufB + (wn + l16) * LDT + g * 16;
      const _Float16* pb1 = bufB + (wn + 16 + l16) * LDT + g * 16;
      #pragma unroll
      for (int j = 0; j < 16; ++j) { b0[j] = pb0[j]; b1[j] = pb1[j]; }
    }
    acc00 = wmma_f16(a0, b0, acc00);
    acc01 = wmma_f16(a0, b1, acc01);
    acc10 = wmma_f16(a1, b0, acc10);
    acc11 = wmma_f16(a1, b1, acc11);
    __syncthreads();                       // reads done before buf reuse
  }

  #pragma unroll
  for (int r = 0; r < 8; ++r) {
    const int m0 = mblk + wm + g * 8 + r;
    const int m1 = m0 + 16;
    C[(size_t)m0 * N + nblk + wn + l16]      = acc00[r];
    C[(size_t)m0 * N + nblk + wn + 16 + l16] = acc01[r];
    C[(size_t)m1 * N + nblk + wn + l16]      = acc10[r];
    C[(size_t)m1 * N + nblk + wn + 16 + l16] = acc11[r];
  }
}

// ---------------------------------------------------------------------------
// RoPE + repack to per-head f16 [h][s][d]; softmax scale folded into Q.
// ---------------------------------------------------------------------------
__global__ __launch_bounds__(256)
void mi455_rope_pack(const float* __restrict__ Qf, const float* __restrict__ Kf,
                     const float* __restrict__ Vf,
                     _Float16* __restrict__ Qh, _Float16* __restrict__ Kh,
                     _Float16* __restrict__ Vh)
{
  const int s = blockIdx.x;
  const float LOGB = 9.210340371976184f;   // ln(10000)
  const float QSCL = 0.08838834764831845f; // 1/sqrt(128)

  for (int idx = threadIdx.x; idx < NH * HD; idx += 256) {
    const int h = idx >> 7, c = idx & 127, i = c & 63;
    const float invf = __expf(-(float)(2 * i) * (1.0f / HD) * LOGB);
    float sn, cs; __sincosf((float)s * invf, &sn, &cs);
    const float x  = Qf[(size_t)s * HID + h * HD + c];
    const float xr = (c < 64) ? -Qf[(size_t)s * HID + h * HD + c + 64]
                              :  Qf[(size_t)s * HID + h * HD + c - 64];
    Qh[(size_t)h * S_LEN * HD + (size_t)s * HD + c] =
        (_Float16)((x * cs + xr * sn) * QSCL);
  }
  for (int idx = threadIdx.x; idx < NKV * HD; idx += 256) {
    const int h = idx >> 7, c = idx & 127, i = c & 63;
    const float invf = __expf(-(float)(2 * i) * (1.0f / HD) * LOGB);
    float sn, cs; __sincosf((float)s * invf, &sn, &cs);
    const float x  = Kf[(size_t)s * KVDIM + h * HD + c];
    const float xr = (c < 64) ? -Kf[(size_t)s * KVDIM + h * HD + c + 64]
                              :  Kf[(size_t)s * KVDIM + h * HD + c - 64];
    Kh[(size_t)h * S_LEN * HD + (size_t)s * HD + c] = (_Float16)(x * cs + xr * sn);
  }
  for (int idx = threadIdx.x; idx < NKV * HD; idx += 256) {
    const int h = idx >> 7, c = idx & 127;
    Vh[(size_t)h * S_LEN * HD + (size_t)s * HD + c] =
        (_Float16)Vf[(size_t)s * KVDIM + h * HD + c];
  }
}

// ---------------------------------------------------------------------------
// Causal flash attention: TDM-staged K/V tiles, online softmax, f16 output.
// ---------------------------------------------------------------------------
#define QB 64
#define KB 32
#define LDP 40    // P scratch row stride (halves)
#define LDKV 132  // 256B data + 8B TDM pad, in halves

__global__ __launch_bounds__(128)
void mi455_flash_attn(const _Float16* __restrict__ Qh, const _Float16* __restrict__ Kh,
                      const _Float16* __restrict__ Vh, _Float16* __restrict__ AO)
{
  __shared__ _Float16 lsP[4][16 * LDP];
  __shared__ _Float16 lsK[2][KB * LDKV];
  __shared__ _Float16 lsV[2][KB * LDKV];

  const int h    = blockIdx.y;
  const int qb   = blockIdx.x;
  const int hkv  = h / GRP;
  const int tid  = threadIdx.x;
  const int w    = tid >> 5;
  const int lane = tid & 31;
  const int g    = lane >> 4;
  const int l16  = lane & 15;

  const int qbase = qb * QB + w * 16;
  const int qrow  = qbase + l16;

  const _Float16* Qp = Qh + (size_t)h   * S_LEN * HD + (size_t)qrow * HD;
  const _Float16* Kp = Kh + (size_t)hkv * S_LEN * HD;
  const _Float16* Vp = Vh + (size_t)hkv * S_LEN * HD;

  // preload Q fragments (d = 128 = 4 chunks of 32)
  v16h aq[4];
  #pragma unroll
  for (int ch = 0; ch < 4; ++ch) {
    #pragma unroll
    for (int j = 0; j < 8; ++j) {
      aq[ch][j]     = Qp[ch * 32 + g * 16 + j];
      aq[ch][j + 8] = Qp[ch * 32 + g * 16 + 8 + j];
    }
  }

  v8f acc[8] = {};
  float mrow[8], lrow[8];
  #pragma unroll
  for (int r = 0; r < 8; ++r) { mrow[r] = -3.0e38f; lrow[r] = 0.0f; }

  const int nkb = qb * 2 + 2;            // identical trip count for all waves
  if (w == 0) {                          // prologue DMA: KV tile 0
    tdm_load_2d_f16(Kp, lds_off_of(&lsK[0][0]), KB, HD, HD, 5);
    tdm_load_2d_f16(Vp, lds_off_of(&lsV[0][0]), KB, HD, HD, 5);
  }

  for (int kb = 0; kb < nkb; ++kb) {
    const int kbase = kb * KB;
    if (w == 0) __builtin_amdgcn_s_wait_tensorcnt(0);
    __syncthreads();                     // KV tile kb resident
    if (kb + 1 < nkb && w == 0) {        // overlap DMA of tile kb+1
      const int nb = (kb + 1) & 1;
      tdm_load_2d_f16(Kp + (size_t)(kbase + KB) * HD, lds_off_of(&lsK[nb][0]),
                      KB, HD, HD, 5);
      tdm_load_2d_f16(Vp + (size_t)(kbase + KB) * HD, lds_off_of(&lsV[nb][0]),
                      KB, HD, HD, 5);
    }
    const _Float16* bufK = &lsK[kb & 1][0];
    const _Float16* bufV = &lsV[kb & 1][0];

    const bool active = (kbase <= qbase + 15);
    if (active) {
      const bool need_mask = (kbase + KB - 1 > qbase);
      v8f s0 = {}, s1 = {};
      #pragma unroll
      for (int ch = 0; ch < 4; ++ch) {
        v16h b0, b1;
        const _Float16* pk0 = bufK + l16 * LDKV        + ch * 32 + g * 16;
        const _Float16* pk1 = bufK + (16 + l16) * LDKV + ch * 32 + g * 16;
        #pragma unroll
        for (int j = 0; j < 16; ++j) { b0[j] = pk0[j]; b1[j] = pk1[j]; }
        s0 = wmma_f16(aq[ch], b0, s0);
        s1 = wmma_f16(aq[ch], b1, s1);
      }
      _Float16* pP = &lsP[w][0];
      #pragma unroll
      for (int r = 0; r < 8; ++r) {
        const int row = g * 8 + r;
        const int q   = qbase + row;
        float v0 = s0[r], v1 = s1[r];
        if (need_mask) {
          if (kbase + l16 > q)      v0 = -3.0e38f;
          if (kbase + 16 + l16 > q) v1 = -3.0e38f;
        }
        float mx = fmaxf(v0, v1);
        #pragma unroll
        for (int off = 1; off < 16; off <<= 1)
          mx = fmaxf(mx, __shfl_xor(mx, off, 32));
        const float mnew = fmaxf(mrow[r], mx);
        const float corr = __expf(mrow[r] - mnew);
        const float p0 = __expf(v0 - mnew);
        const float p1 = __expf(v1 - mnew);
        float ps = p0 + p1;
        #pragma unroll
        for (int off = 1; off < 16; off <<= 1)
          ps += __shfl_xor(ps, off, 32);
        lrow[r] = lrow[r] * corr + ps;
        mrow[r] = mnew;
        #pragma unroll
        for (int nch = 0; nch < 8; ++nch) acc[nch][r] *= corr;
        pP[row * LDP + l16]      = (_Float16)p0;
        pP[row * LDP + 16 + l16] = (_Float16)p1;
      }
    }
    __syncthreads();                     // P visible within wave's LDS region

    if (active) {
      v16h ap;
      const _Float16* pP = &lsP[w][l16 * LDP];
      #pragma unroll
      for (int j = 0; j < 8; ++j) {
        ap[j]     = pP[g * 16 + j];
        ap[j + 8] = pP[g * 16 + 8 + j];
      }
      #pragma unroll
      for (int nch = 0; nch < 8; ++nch) {
        v16h bv;
        const _Float16* pv = bufV + (g * 16) * LDKV + nch * 16 + l16;
        #pragma unroll
        for (int j = 0; j < 16; ++j) bv[j] = pv[j * LDKV];
        acc[nch] = wmma_f16(ap, bv, acc[nch]);
      }
    }
    __syncthreads();                     // KV buf reads done before overwrite
  }

  #pragma unroll
  for (int r = 0; r < 8; ++r) {
    const int row = g * 8 + r;
    const int q   = qbase + row;
    const float inv = 1.0f / lrow[r];
    #pragma unroll
    for (int nch = 0; nch < 8; ++nch)
      AO[(size_t)q * HID + h * HD + nch * 16 + l16] =
          (_Float16)(acc[nch][r] * inv);
  }
}

// ---------------------------------------------------------------------------
extern "C" void kernel_launch(void* const* d_in, const int* in_sizes, int n_in,
                              void* d_out, int out_size, void* d_ws, size_t ws_size,
                              hipStream_t stream)
{
  const float* X  = (const float*)d_in[0];
  const float* Wq = (const float*)d_in[1];
  const float* Wk = (const float*)d_in[2];
  const float* Wv = (const float*)d_in[3];
  const float* Wo = (const float*)d_in[4];
  float* out = (float*)d_out;
  (void)in_sizes; (void)n_in; (void)out_size; (void)ws_size;

  char* ws = (char*)d_ws;
  size_t off = 0;
  _Float16* Xh  = (_Float16*)(ws + off); off += (size_t)S_LEN * HID   * 2;
  _Float16* Wqh = (_Float16*)(ws + off); off += (size_t)HID   * HID   * 2;
  _Float16* Wkh = (_Float16*)(ws + off); off += (size_t)KVDIM * HID   * 2;
  _Float16* Wvh = (_Float16*)(ws + off); off += (size_t)KVDIM * HID   * 2;
  _Float16* Woh = (_Float16*)(ws + off); off += (size_t)HID   * HID   * 2;
  float*    Qf  = (float*)(ws + off);    off += (size_t)S_LEN * HID   * 4;
  float*    Kf  = (float*)(ws + off);    off += (size_t)S_LEN * KVDIM * 4;
  float*    Vf  = (float*)(ws + off);    off += (size_t)S_LEN * KVDIM * 4;
  _Float16* Qh  = (_Float16*)(ws + off); off += (size_t)NH  * S_LEN * HD * 2;
  _Float16* Kh  = (_Float16*)(ws + off); off += (size_t)NKV * S_LEN * HD * 2;
  _Float16* Vh  = (_Float16*)(ws + off); off += (size_t)NKV * S_LEN * HD * 2;
  _Float16* AOh = (_Float16*)(ws + off); off += (size_t)S_LEN * HID * 2;

  const int nX = S_LEN * HID, nWq = HID * HID, nWk = KVDIM * HID;
  mi455_cvt_f16<<<nX  / 1024, 256, 0, stream>>>(X,  Xh,  nX);
  mi455_cvt_f16<<<nWq / 1024, 256, 0, stream>>>(Wq, Wqh, nWq);
  mi455_cvt_f16<<<nWk / 1024, 256, 0, stream>>>(Wk, Wkh, nWk);
  mi455_cvt_f16<<<nWk / 1024, 256, 0, stream>>>(Wv, Wvh, nWk);
  mi455_cvt_f16<<<nWq / 1024, 256, 0, stream>>>(Wo, Woh, nWq);

  mi455_gemm_f16<<<dim3(HID / BN,   S_LEN / BM), 256, 0, stream>>>(Xh, Wqh, Qf, S_LEN, HID,   HID);
  mi455_gemm_f16<<<dim3(KVDIM / BN, S_LEN / BM), 256, 0, stream>>>(Xh, Wkh, Kf, S_LEN, KVDIM, HID);
  mi455_gemm_f16<<<dim3(KVDIM / BN, S_LEN / BM), 256, 0, stream>>>(Xh, Wvh, Vf, S_LEN, KVDIM, HID);
  mi455_rope_pack<<<S_LEN, 256, 0, stream>>>(Qf, Kf, Vf, Qh, Kh, Vh);
  mi455_flash_attn<<<dim3(S_LEN / QB, NH), 128, 0, stream>>>(Qh, Kh, Vh, AOh);
  mi455_gemm_f16<<<dim3(HID / BN, S_LEN / BM), 256, 0, stream>>>(AOh, Woh, out, S_LEN, HID, HID);
}
